// VRWKV_ChannelMix_25598005084646
// MI455X (gfx1250) — matrix operand order repaired and checked by reading, not compiled, not run on hardware
//
#include <hip/hip_runtime.h>
#include <hip/hip_bf16.h>

typedef __bf16 bf16_t;
typedef __attribute__((ext_vector_type(16))) __bf16 v16bf;
typedef __attribute__((ext_vector_type(8)))  __bf16 v8bf;
typedef __attribute__((ext_vector_type(8)))  float  v8f;
typedef __attribute__((ext_vector_type(4)))  int    v4i;

#define C_   192
#define HID_ 768
#define B_   4
#define H_   128
#define W_   128
#define T_   (H_*W_)
#define NROW (B_*T_)            /* 65536 */

/* workspace layout (bytes) */
#define XS_OFF   ((size_t)0)
#define XS_BYTES ((size_t)NROW*C_*2)          /* 25,165,824 */
#define WKF_OFF  (XS_OFF + XS_BYTES)
#define WKF_DW   (48*6*256)                    /* 73728 dwords */
#define WVF_OFF  (WKF_OFF + (size_t)WKF_DW*4)
#define WVF_DW   (24*12*256)                   /* 73728 dwords */
#define WRF_OFF  (WVF_OFF + (size_t)WVF_DW*4)
#define WRF_DW   (12*6*256)                    /* 18432 dwords */
#define KC_OFF   (WRF_OFF + (size_t)WRF_DW*4)

#define WMMA_BF16(A,Bm,Cacc) \
  __builtin_amdgcn_wmma_f32_16x16x32_bf16(false,(A),false,(Bm),(short)0,(Cacc),false,false)

/* CDNA5 async global->LDS path (bypasses VGPRs, tracked by ASYNCcnt). */
#if defined(__has_builtin)
#  if __has_builtin(__builtin_amdgcn_global_load_async_to_lds_b128)
#    define HAVE_ASYNC_LDS 1
#  endif
#endif

typedef __attribute__((address_space(1))) v4i gv4i_t;
typedef __attribute__((address_space(3))) v4i lv4i_t;

__device__ __forceinline__ void async_copy16(const void* gsrc, void* lds_dst) {
#ifdef HAVE_ASYNC_LDS
  __builtin_amdgcn_global_load_async_to_lds_b128(
      (gv4i_t*)gsrc, (lv4i_t*)lds_dst, 0, 0);
#else
  *(uint4*)lds_dst = *(const uint4*)gsrc;
#endif
}

__device__ __forceinline__ void async_wait() {
#ifdef HAVE_ASYNC_LDS
  asm volatile("s_wait_asynccnt 0x0" ::: "memory");
#endif
}

__device__ __forceinline__ v16bf combine16(v8bf lo, v8bf hi) {
  return __builtin_shufflevector(lo, hi, 0,1,2,3,4,5,6,7,8,9,10,11,12,13,14,15);
}

/* -------- prep 1: fold 1x1 + 3x3 + 5x5 + identity into one 5x5 kernel -------- */
__global__ void kcomb_prep(const float* __restrict__ w1, const float* __restrict__ w3,
                           const float* __restrict__ w5, const float* __restrict__ alpha,
                           float* __restrict__ kc) {
  int idx = blockIdx.x * 256 + threadIdx.x;
  if (idx >= C_ * 25) return;
  int c = idx / 25, j = idx % 25;
  int dy = j / 5 - 2, dx = j % 5 - 2;
  float v = alpha[3] * w5[idx];
  if (dy >= -1 && dy <= 1 && dx >= -1 && dx <= 1)
    v += alpha[2] * w3[c * 9 + (dy + 1) * 3 + (dx + 1)];
  if (dy == 0 && dx == 0)
    v += alpha[1] * w1[c] + alpha[0];
  kc[idx] = v;
}

/* -------- prep 2: pre-swizzle Wk/Wv/Wr into WMMA B-fragment layout (bf16) ----
   B 32x16 frag: lane l -> column N = l%16 ; K = (l>=16)*8 + slot (slot<8), 16 + (l>=16)*8 + slot-8 (slot>=8).
   Stored as [frag][lane][8 dwords] so the GEMM loads 32B/lane flat.            */
__global__ void frag_prep(const float* __restrict__ Wk, const float* __restrict__ Wr,
                          const float* __restrict__ Wv, unsigned* __restrict__ wkf,
                          unsigned* __restrict__ wvf, unsigned* __restrict__ wrf) {
  int idx = blockIdx.x * 256 + threadIdx.x;
  if (idx >= WKF_DW + WVF_DW + WRF_DW) return;
  const float* src; unsigned* dst; int local, mode;
  if (idx < WKF_DW)                 { mode = 0; local = idx;                    dst = wkf; src = Wk; }
  else if (idx < WKF_DW + WVF_DW)   { mode = 1; local = idx - WKF_DW;           dst = wvf; src = Wv; }
  else                              { mode = 2; local = idx - WKF_DW - WVF_DW;  dst = wrf; src = Wr; }
  int f    = local >> 8;           /* fragment index            */
  int r    = local & 255;
  int lane = r >> 3;
  int v    = r & 7;
  int N    = lane & 15;
  int Kb   = ((lane >> 4) << 3) + 2 * v + ((v >> 2) << 3);
  const float* p;
  if (mode == 0) {            /* GEMM1: B[K=c][N=h] = Wk[h,c] ; frag [ht][kc] */
    int ht = f / 6, kc = f % 6;
    p = src + (size_t)(ht * 16 + N) * C_ + kc * 32 + Kb;
  } else if (mode == 1) {     /* GEMM2: B[K=h][N=c] = Wv[c,h] ; frag [hc][ct] */
    int hc = f / 12, ct = f % 12;
    p = src + (size_t)(ct * 16 + N) * HID_ + hc * 32 + Kb;
  } else {                    /* GEMM3: B[K=c][N=d] = Wr[d,c] ; frag [ct][kc] */
    int ct = f / 6, kc = f % 6;
    p = src + (size_t)(ct * 16 + N) * C_ + kc * 32 + Kb;
  }
  union { bf16_t b[2]; unsigned u; } pk;
  pk.b[0] = (bf16_t)p[0];
  pk.b[1] = (bf16_t)p[1];
  dst[local] = pk.u;
}

/* -------- stencil: xs = folded 5x5 depthwise conv, store bf16 [N,C] -------- */
__global__ __launch_bounds__(256) void stencil_kernel(const float* __restrict__ x,
                                                      const float* __restrict__ kc,
                                                      bf16_t* __restrict__ xs) {
  int idx = blockIdx.x * 256 + threadIdx.x;      /* over NROW*C */
  int c = idx % C_;
  int t = (idx / C_) & (T_ - 1);
  int b = idx / (C_ * T_);
  int h = t >> 7, w = t & (W_ - 1);
  const float* xb  = x + (size_t)b * T_ * C_ + c;
  const float* kcp = kc + c * 25;
  float acc = 0.f;
#pragma unroll
  for (int dy = -2; dy <= 2; ++dy) {
    int hh = h + dy;
    if ((unsigned)hh < (unsigned)H_) {
#pragma unroll
      for (int dx = -2; dx <= 2; ++dx) {
        int ww = w + dx;
        if ((unsigned)ww < (unsigned)W_)
          acc += kcp[(dy + 2) * 5 + dx + 2] * xb[(size_t)(hh * W_ + ww) * C_];
      }
    }
  }
  xs[idx] = (bf16_t)acc;
}

/* -------- fused GEMM: kv = relu^2(xs Wk^T) Wv^T ; out = sigmoid(xs Wr^T)*kv --
   8 waves/WG, 16 rows/wave. Per hid-chunk (32): double-buffered async staging
   of 12 Wk + 12 Wv B-frags (one barrier per chunk), 12 WMMAs -> k-chunk,
   relu^2 -> bf16 -> LDS relayout (D-frag -> A-frag), 12 WMMAs into kv
   accumulators (96 VGPRs/lane).                                               */
__global__ __launch_bounds__(256) void gemm_fused(const bf16_t* __restrict__ xs,
                                                  const unsigned* __restrict__ wkf,
                                                  const unsigned* __restrict__ wvf,
                                                  const unsigned* __restrict__ wrf,
                                                  float* __restrict__ out) {
  __shared__ __align__(32) unsigned stage[2][6144];       /* 2 x 24 KB staging      */
  __shared__ __align__(32) bf16_t   ldsk[8 * 16 * 40];    /* 10 KB, 80B row stride  */
  const int tid    = threadIdx.x;
  const int lane   = tid & 31;
  const int wid    = tid >> 5;
  const int row0   = blockIdx.x * 128 + wid * 16;
  const int mrow   = lane & 15;
  const int hiHalf = lane >> 4;
  const int baseA  = hiHalf * 8;     /* A-frag K base per ISA 16-bit A layout */
  const int N      = lane & 15;      /* D-frag column                          */
  const int Mbase  = hiHalf * 8;     /* D-frag row base                        */

  /* A fragments of xs: lane = row mrow, K runs [base,base+8) & [base+16,base+24) */
  const bf16_t* xrow = xs + (size_t)(row0 + mrow) * C_;
  v16bf axs[6];
#pragma unroll
  for (int kc = 0; kc < 6; ++kc) {
    v8bf lo = *(const v8bf*)(xrow + kc * 32 + baseA);
    v8bf hi = *(const v8bf*)(xrow + kc * 32 + baseA + 16);
    axs[kc] = combine16(lo, hi);
  }

  v8f kv[12];
#pragma unroll
  for (int ct = 0; ct < 12; ++ct) kv[ct] = (v8f){0.f,0.f,0.f,0.f,0.f,0.f,0.f,0.f};

  bf16_t* kbase = ldsk + wid * (16 * 40);

  /* stage chunk 0 */
  {
    const uint4* srcK = (const uint4*)(wkf);
    const uint4* srcV = (const uint4*)(wvf);
    uint4* dst = (uint4*)stage[0];
#pragma unroll
    for (int i = 0; i < 3; ++i) async_copy16(&srcK[tid + i * 256], &dst[tid + i * 256]);
#pragma unroll
    for (int i = 0; i < 3; ++i) async_copy16(&srcV[tid + i * 256], &dst[768 + tid + i * 256]);
  }
  async_wait();
  __syncthreads();

  for (int hc = 0; hc < 24; ++hc) {
    const char* buf = (const char*)stage[hc & 1];

    /* prefetch next chunk's fragments into the other buffer (async, no VGPRs) */
    if (hc + 1 < 24) {
      const uint4* srcK = (const uint4*)(wkf + (hc + 1) * 3072);
      const uint4* srcV = (const uint4*)(wvf + (hc + 1) * 3072);
      uint4* dst = (uint4*)stage[(hc + 1) & 1];
#pragma unroll
      for (int i = 0; i < 3; ++i) async_copy16(&srcK[tid + i * 256], &dst[tid + i * 256]);
#pragma unroll
      for (int i = 0; i < 3; ++i) async_copy16(&srcV[tid + i * 256], &dst[768 + tid + i * 256]);
    }

    /* GEMM1: two 16x16 k-tiles for this 32-wide hid chunk */
    v8f a0 = (v8f){0.f,0.f,0.f,0.f,0.f,0.f,0.f,0.f};
    v8f a1 = (v8f){0.f,0.f,0.f,0.f,0.f,0.f,0.f,0.f};
#pragma unroll
    for (int kc = 0; kc < 6; ++kc) {
      v16bf b0 = *(const v16bf*)(buf + (size_t)kc * 1024 + lane * 32);
      v16bf b1 = *(const v16bf*)(buf + (size_t)(6 + kc) * 1024 + lane * 32);
      a0 = WMMA_BF16(axs[kc], b0, a0);
      a1 = WMMA_BF16(axs[kc], b1, a1);
    }

    /* relu^2 -> bf16 -> LDS (row-major 16x32, stride 40 = no bank conflicts) */
#pragma unroll
    for (int i = 0; i < 8; ++i) {
      float f0 = fmaxf(a0[i], 0.f); f0 *= f0;
      float f1 = fmaxf(a1[i], 0.f); f1 *= f1;
      kbase[(Mbase + i) * 40 + N]      = (bf16_t)f0;
      kbase[(Mbase + i) * 40 + 16 + N] = (bf16_t)f1;
    }
    asm volatile("s_wait_dscnt 0x0" ::: "memory");  /* same-wave DS is in-order */

    /* re-read as A fragment of k-chunk */
    const bf16_t* krow = kbase + mrow * 40;
    v8bf klo = *(const v8bf*)(krow + baseA);
    v8bf khi = *(const v8bf*)(krow + baseA + 16);
    v16bf ka = combine16(klo, khi);

    /* GEMM2: accumulate kv over all 12 output-channel tiles */
#pragma unroll
    for (int ct = 0; ct < 12; ++ct) {
      v16bf bv = *(const v16bf*)(buf + 12288 + (size_t)ct * 1024 + lane * 32);
      kv[ct] = WMMA_BF16(ka, bv, kv[ct]);
    }

    async_wait();       /* next chunk's async stage complete            */
    __syncthreads();    /* all waves done reading buf before its reuse  */
  }

  /* GEMM3 + epilogue: r = sigmoid(xs Wr^T) ; out = r * kv */
#pragma unroll
  for (int ct = 0; ct < 12; ++ct) {
    v8f rr = (v8f){0.f,0.f,0.f,0.f,0.f,0.f,0.f,0.f};
#pragma unroll
    for (int kc = 0; kc < 6; ++kc) {
      v16bf br = *(const v16bf*)(wrf + (size_t)(ct * 6 + kc) * 256 + lane * 8);
      rr = WMMA_BF16(axs[kc], br, rr);
    }
#pragma unroll
    for (int i = 0; i < 8; ++i) {
      float sg = 1.0f / (1.0f + __expf(-rr[i]));
      out[(size_t)(row0 + Mbase + i) * C_ + ct * 16 + N] = sg * kv[ct][i];
    }
  }
}

extern "C" void kernel_launch(void* const* d_in, const int* in_sizes, int n_in,
                              void* d_out, int out_size, void* d_ws, size_t ws_size,
                              hipStream_t stream) {
  const float* x     = (const float*)d_in[0];
  const float* w1    = (const float*)d_in[1];
  const float* w3    = (const float*)d_in[2];
  const float* w5    = (const float*)d_in[3];
  const float* alpha = (const float*)d_in[4];
  const float* Wk    = (const float*)d_in[5];
  const float* Wr    = (const float*)d_in[6];
  const float* Wv    = (const float*)d_in[7];
  float* out = (float*)d_out;

  char* ws = (char*)d_ws;
  bf16_t*   xs  = (bf16_t*)(ws + XS_OFF);
  unsigned* wkf = (unsigned*)(ws + WKF_OFF);
  unsigned* wvf = (unsigned*)(ws + WVF_OFF);
  unsigned* wrf = (unsigned*)(ws + WRF_OFF);
  float*    kc  = (float*)(ws + KC_OFF);

  kcomb_prep<<<(C_ * 25 + 255) / 256, 256, 0, stream>>>(w1, w3, w5, alpha, kc);
  frag_prep<<<(WKF_DW + WVF_DW + WRF_DW + 255) / 256, 256, 0, stream>>>(Wk, Wr, Wv, wkf, wvf, wrf);
  stencil_kernel<<<(NROW * C_) / 256, 256, 0, stream>>>(x, kc, xs);
  gemm_fused<<<NROW / 128, 256, 0, stream>>>(xs, wkf, wvf, wrf, out);
}
